// RAFTOpticalFlow_46205258170981
// MI455X (gfx1250) — compile-verified, matrix operand order checked by reading, not run on hardware
//
#include <hip/hip_runtime.h>
#include <hip/hip_bf16.h>
#include <math.h>

// ---------------------------------------------------------------------------
// RAFT-small forward for MI455X (gfx1250, wave32).
//  - All-pairs correlation: blocked WMMA GEMM, f16 operands pre-packed into
//    the ISA 16x16x32 register layout; B tile staged in LDS via the Tensor
//    Data Mover (TENSOR_LOAD_TO_LDS + s_wait_tensorcnt), 12 WMMAs/wave.
//  - 1x1 motion conv (196->96): WMMA GEMM with zero-padded K (196->224),
//    fused bias+ReLU epilogue.
//  - GRU / 3x3 convs: direct fp32 VALU (accuracy-sensitive recurrence).
// ---------------------------------------------------------------------------

typedef __attribute__((ext_vector_type(16))) _Float16 v16h;
typedef __attribute__((ext_vector_type(8)))  float    v8f;
typedef unsigned int u32x4 __attribute__((ext_vector_type(4)));
typedef int          i32x4 __attribute__((ext_vector_type(4)));
typedef int          i32x8 __attribute__((ext_vector_type(8)));

#if defined(__has_builtin)
#  if __has_builtin(__builtin_amdgcn_tensor_load_to_lds) && \
      __has_builtin(__builtin_amdgcn_s_wait_tensorcnt)
#    define RAFT_HAVE_TDM 1
#  endif
#endif
#ifndef RAFT_HAVE_TDM
#  define RAFT_HAVE_TDM 0
#endif

// WMMA 16-bit operand element->K mapping (ISA 7.12.2, 16-bit A 16x32):
//   lane<16 : e0..7 -> K 0..7,  e8..15 -> K 16..23
//   lane>=16: e0..7 -> K 8..15, e8..15 -> K 24..31
__device__ __forceinline__ int wmma_klocal(int e, int lane) {
  return (e & 7) + ((e & 8) << 1) + ((lane >> 4) << 3);
}

// ---------------- generic direct conv (supports 2 concatenated inputs) -----
// act: 0=none 1=relu 2=sigmoid 3=tanh ; pre: 1 => input transform 2*v-1
__global__ void conv2d_kernel(const float* __restrict__ inA, int Ca,
                              const float* __restrict__ inB, int Cb,
                              const float* __restrict__ wgt,
                              const float* __restrict__ bias,
                              float* __restrict__ out,
                              int Bn, int H, int W, int Co,
                              int K, int stride, int pad,
                              int OH, int OW, int act, int pre) {
  int idx = blockIdx.x * blockDim.x + threadIdx.x;
  int total = Bn * Co * OH * OW;
  if (idx >= total) return;
  int ox = idx % OW; int t = idx / OW;
  int oy = t % OH;   t /= OH;
  int co = t % Co;   int b = t / Co;
  int Ci = Ca + Cb;
  float sum = bias[co];
  int iy0 = oy * stride - pad;
  int ix0 = ox * stride - pad;
  for (int ci = 0; ci < Ci; ++ci) {
    const float* src = (ci < Ca)
        ? (inA + ((size_t)b * Ca + ci) * (size_t)H * W)
        : (inB + ((size_t)b * Cb + (ci - Ca)) * (size_t)H * W);
    const float* wp = wgt + ((size_t)co * Ci + ci) * (size_t)K * K;
    for (int ky = 0; ky < K; ++ky) {
      int iy = iy0 + ky;
      if (iy < 0 || iy >= H) continue;
      for (int kx = 0; kx < K; ++kx) {
        int ix = ix0 + kx;
        if (ix < 0 || ix >= W) continue;
        float v = src[(size_t)iy * W + ix];
        if (pre) v = 2.0f * v - 1.0f;
        sum += v * wp[ky * K + kx];
      }
    }
  }
  if (act == 1)      sum = fmaxf(sum, 0.0f);
  else if (act == 2) sum = 1.0f / (1.0f + expf(-sum));
  else if (act == 3) sum = tanhf(sum);
  out[idx] = sum;
}

// ---------------- operand packing into WMMA register layout ----------------
// src: [Bn, Csrc, HW] fp32 (row-major).  dst element index:
//   (((b*NT + tile)*KS + ks)*32 + lane)*16 + e
// column = tile*16 + (lane&15), channel = ks*32 + wmma_klocal(e,lane);
// zero-padded for channel >= Csrc.
__global__ void pack_rhs_kernel(const float* __restrict__ src,
                                _Float16* __restrict__ dst,
                                int Bn, int Csrc, int KS, int NT, int HW) {
  long long idx = (long long)blockIdx.x * blockDim.x + threadIdx.x;
  long long total = (long long)Bn * NT * KS * 512;
  if (idx >= total) return;
  int e    = (int)(idx & 15);
  int lane = (int)((idx >> 4) & 31);
  long long t = idx >> 9;
  int ks   = (int)(t % KS); t /= KS;
  int tile = (int)(t % NT);
  int b    = (int)(t / NT);
  int c    = ks * 32 + wmma_klocal(e, lane);
  int col  = tile * 16 + (lane & 15);
  float v = 0.0f;
  if (c < Csrc) v = src[((size_t)b * Csrc + c) * HW + col];
  dst[idx] = (_Float16)v;
}

// weight pack: w [Co, Csrc] -> A-layout [(Co/16) tiles, KS, 32 lanes, 16 e]
__global__ void pack_w_kernel(const float* __restrict__ w,
                              _Float16* __restrict__ dst,
                              int Co, int Csrc, int KS) {
  int idx = blockIdx.x * blockDim.x + threadIdx.x;
  int total = (Co / 16) * KS * 512;
  if (idx >= total) return;
  int e    = idx & 15;
  int lane = (idx >> 4) & 31;
  int t    = idx >> 9;
  int ks   = t % KS;
  int mt   = t / KS;
  int c    = ks * 32 + wmma_klocal(e, lane);
  int m    = mt * 16 + (lane & 15);
  float v = 0.0f;
  if (c < Csrc) v = w[(size_t)m * Csrc + c];
  dst[idx] = (_Float16)v;
}

// ---------------- blocked all-pairs correlation GEMM -----------------------
// 3 waves/workgroup -> 48(M) x 48(N) tile; B tile (3 n-tiles x 4 k-steps,
// 12KB contiguous) staged in LDS by the Tensor Data Mover and shared by all
// waves; A held in registers.  grid = (HW/48, HW/48, Bn), block = 96.
__global__ __launch_bounds__(96)
void corr_gemm_kernel(const _Float16* __restrict__ apack,
                      const _Float16* __restrict__ bpack,
                      float* __restrict__ corr, int HW, float scale) {
  __shared__ __align__(32) uint4 ldsb[768];   // 3*4*32 v16h = 12KB
  const int tid  = threadIdx.x;               // 0..95
  const int wave = tid >> 5;                  // 0..2
  const int lane = tid & 31;
  const int NT   = HW / 16;                   // 225
  const int nt0  = blockIdx.x * 3;
  const int mtile = blockIdx.y * 3 + wave;
  const int b    = blockIdx.z;

  const uint4* bsrc = (const uint4*)bpack + (size_t)(b * NT + nt0) * 256;
  const v16h* av = (const v16h*)apack + (size_t)(b * NT + mtile) * 4 * 32;
  __builtin_prefetch((const void*)av, 0, 1);  // global_prefetch_b8 on A stream

#if RAFT_HAVE_TDM
  if (wave == 0) {
    // Tensor DMA descriptor (ISA ch.8): 1-D tile, 3072 x 4B = 12KB.
    unsigned ldsOff = (unsigned)(uintptr_t)&ldsb[0];   // low 32b of generic
    unsigned long long ga = (unsigned long long)bsrc;  // LDS ptr = LDS offset
    u32x4 g0 = { 1u,                                   // count=1 (user D#)
                 ldsOff,                                // lds_addr
                 (unsigned)(ga & 0xFFFFFFFFu),          // global_addr[31:0]
                 (unsigned)((ga >> 32) & 0x01FFFFFFu)   // global_addr[56:32]
                   | (2u << 30) };                      // type=2 ("image")
    i32x8 g1 = { (int)(2u << 16),      // data_size=2 (4 bytes)
                 (int)(3072u << 16),   // tensor_dim0[15:0]
                 (int)(1u << 16),      // tensor_dim0[31:16]=0, tensor_dim1=1
                 (int)(3072u << 16),   // tensor_dim1[31:16]=0, tile_dim0=3072
                 0,                    // tile_dim1=0 (1-D), tile_dim2=0
                 3072,                 // tensor_dim0_stride[31:0]
                 0, 0 };               // stride hi / dim1 stride
    i32x4 gz = { 0, 0, 0, 0 };
#if __clang_major__ >= 23
    i32x8 gz8 = { 0, 0, 0, 0, 0, 0, 0, 0 };
    __builtin_amdgcn_tensor_load_to_lds(g0, g1, gz, gz, gz8, 0);
#else
    __builtin_amdgcn_tensor_load_to_lds(g0, g1, gz, gz, 0);
#endif
    __builtin_amdgcn_s_wait_tensorcnt(0);
  }
  __syncthreads();
#else
  for (int i = tid; i < 768; i += 96) ldsb[i] = bsrc[i];
  __syncthreads();
#endif

  // A: this wave's m-tile, 4 k-step vectors in registers
  v16h a0 = av[0 * 32 + lane];
  v16h a1 = av[1 * 32 + lane];
  v16h a2 = av[2 * 32 + lane];
  v16h a3 = av[3 * 32 + lane];

  const v16h* bl = (const v16h*)ldsb;
  const int lmod  = lane & 15;
  const int mbase = mtile * 16 + ((lane >> 4) << 3);
  float* out = corr + (size_t)b * HW * HW;

#pragma unroll
  for (int j = 0; j < 3; ++j) {
    v16h b0 = bl[(j * 4 + 0) * 32 + lane];
    v16h b1 = bl[(j * 4 + 1) * 32 + lane];
    v16h b2 = bl[(j * 4 + 2) * 32 + lane];
    v16h b3 = bl[(j * 4 + 3) * 32 + lane];
    v8f acc = {};
    acc = __builtin_amdgcn_wmma_f32_16x16x32_f16(false, a0, false, b0, (short)0, acc, false, false);
    acc = __builtin_amdgcn_wmma_f32_16x16x32_f16(false, a1, false, b1, (short)0, acc, false, false);
    acc = __builtin_amdgcn_wmma_f32_16x16x32_f16(false, a2, false, b2, (short)0, acc, false, false);
    acc = __builtin_amdgcn_wmma_f32_16x16x32_f16(false, a3, false, b3, (short)0, acc, false, false);
    const int n = (nt0 + j) * 16 + lmod;
#pragma unroll
    for (int r = 0; r < 8; ++r) {
      out[(size_t)(mbase + r) * HW + n] = acc[r] * scale;
    }
  }
}

// ---------------- 1x1 conv (196->96) as WMMA GEMM, bias+ReLU epilogue ------
// grid = (HW/16, Co/16, Bn), block = 32 (one wave per 16x16 tile).
__global__ __launch_bounds__(32)
void conv1x1_wmma_kernel(const _Float16* __restrict__ wpack,
                         const _Float16* __restrict__ xpack,
                         const float* __restrict__ bias,
                         float* __restrict__ out,
                         int Co, int KS, int HW) {
  const int lane = threadIdx.x;
  const int nt = blockIdx.x;
  const int mt = blockIdx.y;
  const int b  = blockIdx.z;
  const int NT = HW / 16;
  const v16h* av = (const v16h*)wpack + (size_t)mt * KS * 32;
  const v16h* bv = (const v16h*)xpack + (size_t)(b * NT + nt) * KS * 32;
  v8f acc = {};
  for (int ks = 0; ks < KS; ++ks) {
    acc = __builtin_amdgcn_wmma_f32_16x16x32_f16(
        false, av[ks * 32 + lane], false, bv[ks * 32 + lane],
        (short)0, acc, false, false);
  }
  const int lmod  = lane & 15;
  const int mbase = mt * 16 + ((lane >> 4) << 3);
  const int n = nt * 16 + lmod;
#pragma unroll
  for (int r = 0; r < 8; ++r) {
    int m = mbase + r;
    float v = acc[r] + bias[m];
    out[((size_t)b * Co + m) * HW + n] = fmaxf(v, 0.0f);
  }
}

// ---------------- 2x2 avg pool (VALID) over [N, H, W] planes ---------------
__global__ void avgpool2_kernel(const float* __restrict__ in, float* __restrict__ out,
                                int N, int H, int W, int OH, int OW) {
  int idx = blockIdx.x * blockDim.x + threadIdx.x;
  int total = N * OH * OW;
  if (idx >= total) return;
  int ox = idx % OW; int t = idx / OW;
  int oy = t % OH;   int n = t / OH;
  const float* p = in + (size_t)n * H * W;
  int iy = oy * 2, ix = ox * 2;
  float s = p[(size_t)iy * W + ix] + p[(size_t)iy * W + ix + 1]
          + p[(size_t)(iy + 1) * W + ix] + p[(size_t)(iy + 1) * W + ix + 1];
  out[idx] = 0.25f * s;
}

// ---------------- correlation pyramid lookup -------------------------------
__global__ void corr_lookup_kernel(const float* __restrict__ p0,
                                   const float* __restrict__ p1,
                                   const float* __restrict__ p2,
                                   const float* __restrict__ p3,
                                   const float* __restrict__ coords,
                                   float* __restrict__ cf,
                                   int Bn, int h, int w) {
  int idx = blockIdx.x * blockDim.x + threadIdx.x;
  int HW = h * w;
  int total = Bn * 196 * HW;
  if (idx >= total) return;
  int x = idx % w; int t = idx / w;
  int y = t % h;   t /= h;
  int ch = t % 196; int b = t / 196;
  int l = ch / 49;
  int k = ch % 49;
  float cx = coords[(((size_t)b * 2 + 0) * h + y) * w + x];
  float cy = coords[(((size_t)b * 2 + 1) * h + y) * w + x];
  const float* base; int hl, wl;
  if (l == 0)      { base = p0; hl = 45; wl = 80; }
  else if (l == 1) { base = p1; hl = 22; wl = 40; }
  else if (l == 2) { base = p2; hl = 11; wl = 20; }
  else             { base = p3; hl = 5;  wl = 10; }
  float inv = 1.0f / (float)(1 << l);
  float dx = (float)(k % 7) - 3.0f;
  float dy = (float)(k / 7) - 3.0f;
  float xs = cx * inv + dx;
  float ys = cy * inv + dy;
  const float* plane = base + (size_t)(b * HW + y * w + x) * (size_t)(hl * wl);
  float x0f = floorf(xs), y0f = floorf(ys);
  float wx = xs - x0f,    wy = ys - y0f;
  int x0 = (int)x0f, y0 = (int)y0f;
  float v00 = 0.f, v01 = 0.f, v10 = 0.f, v11 = 0.f;
  bool xa = (x0 >= 0 && x0 < wl), xb = (x0 + 1 >= 0 && x0 + 1 < wl);
  bool ya = (y0 >= 0 && y0 < hl), yb = (y0 + 1 >= 0 && y0 + 1 < hl);
  if (ya && xa) v00 = plane[(size_t)y0 * wl + x0];
  if (ya && xb) v01 = plane[(size_t)y0 * wl + x0 + 1];
  if (yb && xa) v10 = plane[(size_t)(y0 + 1) * wl + x0];
  if (yb && xb) v11 = plane[(size_t)(y0 + 1) * wl + x0 + 1];
  cf[idx] = v00 * (1 - wx) * (1 - wy) + v01 * wx * (1 - wy)
          + v10 * (1 - wx) * wy + v11 * wx * wy;
}

// ---------------- small elementwise kernels --------------------------------
__global__ void split_cn_kernel(const float* __restrict__ cn,
                                float* __restrict__ hdn, float* __restrict__ ctx,
                                int Bn, int hw) {
  int idx = blockIdx.x * blockDim.x + threadIdx.x;
  int total = Bn * 160 * hw;
  if (idx >= total) return;
  int p = idx % hw; int t = idx / hw;
  int c = t % 160;  int b = t / 160;
  float v = cn[idx];
  if (c < 96) hdn[((size_t)b * 96 + c) * hw + p] = tanhf(v);
  else        ctx[((size_t)b * 64 + (c - 96)) * hw + p] = fmaxf(v, 0.0f);
}

__global__ void set_coords_kernel(float* __restrict__ coords, int Bn, int h, int w) {
  int idx = blockIdx.x * blockDim.x + threadIdx.x;
  int total = Bn * 2 * h * w;
  if (idx >= total) return;
  int x = idx % w; int t = idx / w;
  int y = t % h;   t /= h;
  int c = t % 2;
  coords[idx] = (c == 0) ? (float)x : (float)y;
}

__global__ void build_flow_kernel(const float* __restrict__ coords,
                                  float* __restrict__ flow, int Bn, int h, int w) {
  int idx = blockIdx.x * blockDim.x + threadIdx.x;
  int total = Bn * 2 * h * w;
  if (idx >= total) return;
  int x = idx % w; int t = idx / w;
  int y = t % h;   t /= h;
  int c = t % 2;
  flow[idx] = coords[idx] - ((c == 0) ? (float)x : (float)y);
}

// xin = [ctx(64), mo(80), flow(2)]  (146 channels)
__global__ void build_xin_kernel(const float* __restrict__ ctx,
                                 const float* __restrict__ mo,
                                 const float* __restrict__ coords,
                                 float* __restrict__ xin, int Bn, int h, int w) {
  int idx = blockIdx.x * blockDim.x + threadIdx.x;
  int hw = h * w;
  int total = Bn * 146 * hw;
  if (idx >= total) return;
  int x = idx % w; int t = idx / w;
  int y = t % h;   t /= h;
  int ch = t % 146; int b = t / 146;
  float v;
  if (ch < 64)        v = ctx[((size_t)b * 64 + ch) * hw + y * w + x];
  else if (ch < 144)  v = mo[((size_t)b * 80 + (ch - 64)) * hw + y * w + x];
  else {
    int c = ch - 144;
    v = coords[(((size_t)b * 2 + c) * h + y) * w + x] - ((c == 0) ? (float)x : (float)y);
  }
  xin[idx] = v;
}

__global__ void mul_kernel(const float* __restrict__ a, const float* __restrict__ b,
                           float* __restrict__ o, int n) {
  int i = blockIdx.x * blockDim.x + threadIdx.x;
  if (i < n) o[i] = a[i] * b[i];
}

__global__ void gru_update_kernel(float* __restrict__ hdn, const float* __restrict__ z,
                                  const float* __restrict__ q, int n) {
  int i = blockIdx.x * blockDim.x + threadIdx.x;
  if (i < n) hdn[i] = (1.0f - z[i]) * hdn[i] + z[i] * q[i];
}

__global__ void add_inplace_kernel(float* __restrict__ a, const float* __restrict__ b, int n) {
  int i = blockIdx.x * blockDim.x + threadIdx.x;
  if (i < n) a[i] += b[i];
}

// ---------------- final flow upsample (half-pixel bilinear, x8, scale 8) ---
__global__ void upsample_flow_kernel(const float* __restrict__ coords,
                                     float* __restrict__ out,
                                     int Bn, int h, int w, int H, int W) {
  int idx = blockIdx.x * blockDim.x + threadIdx.x;
  int total = Bn * 2 * H * W;
  if (idx >= total) return;
  int X = idx % W; int t = idx / W;
  int Y = t % H;   t /= H;
  int c = t % 2;   int b = t / 2;
  float sx = (X + 0.5f) / 8.0f - 0.5f;
  float sy = (Y + 0.5f) / 8.0f - 0.5f;
  float x0f = floorf(sx), y0f = floorf(sy);
  float wx = sx - x0f,    wy = sy - y0f;
  int x0 = (int)x0f, y0 = (int)y0f;
  int x0c = min(max(x0, 0), w - 1);
  int x1c = min(max(x0 + 1, 0), w - 1);
  int y0c = min(max(y0, 0), h - 1);
  int y1c = min(max(y0 + 1, 0), h - 1);
  const float* cc = coords + (((size_t)b * 2 + c) * h) * w;
  float f00 = cc[(size_t)y0c * w + x0c] - ((c == 0) ? (float)x0c : (float)y0c);
  float f01 = cc[(size_t)y0c * w + x1c] - ((c == 0) ? (float)x1c : (float)y0c);
  float f10 = cc[(size_t)y1c * w + x0c] - ((c == 0) ? (float)x0c : (float)y1c);
  float f11 = cc[(size_t)y1c * w + x1c] - ((c == 0) ? (float)x1c : (float)y1c);
  float v = f00 * (1 - wx) * (1 - wy) + f01 * wx * (1 - wy)
          + f10 * (1 - wx) * wy + f11 * wx * wy;
  out[idx] = 8.0f * v;
}

// ---------------------------------------------------------------------------
extern "C" void kernel_launch(void* const* d_in, const int* in_sizes, int n_in,
                              void* d_out, int out_size, void* d_ws, size_t ws_size,
                              hipStream_t stream) {
  (void)in_sizes; (void)n_in; (void)out_size; (void)ws_size;

  const float* img1 = (const float*)d_in[0];
  const float* img2 = (const float*)d_in[1];
  const float* f1_w = (const float*)d_in[2];  const float* f1_b = (const float*)d_in[3];
  const float* f2_w = (const float*)d_in[4];  const float* f2_b = (const float*)d_in[5];
  const float* f3_w = (const float*)d_in[6];  const float* f3_b = (const float*)d_in[7];
  const float* c1_w = (const float*)d_in[8];  const float* c1_b = (const float*)d_in[9];
  const float* c2_w = (const float*)d_in[10]; const float* c2_b = (const float*)d_in[11];
  const float* c3_w = (const float*)d_in[12]; const float* c3_b = (const float*)d_in[13];
  const float* mc1_w = (const float*)d_in[14]; const float* mc1_b = (const float*)d_in[15];
  const float* mf1_w = (const float*)d_in[16]; const float* mf1_b = (const float*)d_in[17];
  const float* mf2_w = (const float*)d_in[18]; const float* mf2_b = (const float*)d_in[19];
  const float* mo_w  = (const float*)d_in[20]; const float* mo_b  = (const float*)d_in[21];
  const float* gz_w  = (const float*)d_in[22]; const float* gz_b  = (const float*)d_in[23];
  const float* gr_w  = (const float*)d_in[24]; const float* gr_b  = (const float*)d_in[25];
  const float* gq_w  = (const float*)d_in[26]; const float* gq_b  = (const float*)d_in[27];
  const float* h1_w  = (const float*)d_in[28]; const float* h1_b  = (const float*)d_in[29];
  const float* h2_w  = (const float*)d_in[30]; const float* h2_b  = (const float*)d_in[31];

  constexpr int Bn = 2, H = 360, W = 640;
  constexpr int H2 = 180, W2 = 320, H4 = 90, W4 = 160;
  constexpr int h = 45, w = 80, HW = h * w;            // 3600, = 16*225 = 48*75
  constexpr int C = 128;
  constexpr int ITERS = 8;
  constexpr int NT = HW / 16;                          // 225 column tiles

  // workspace carve-out (all chunks multiples of 8 floats -> 32B alignment)
  float* ws = (float*)d_ws;
  size_t off = 0;
  auto A = [&](size_t n) { float* p = ws + off; off += n; return p; };
  float* t1    = A((size_t)Bn * 32 * H2 * W2);
  float* t2    = A((size_t)Bn * 64 * H4 * W4);
  float* f1f   = A((size_t)Bn * C * HW);
  float* f2f   = A((size_t)Bn * C * HW);
  float* cn    = A((size_t)Bn * 160 * HW);
  float* hdn   = A((size_t)Bn * 96 * HW);
  float* ctx   = A((size_t)Bn * 64 * HW);
  float* corr0 = A((size_t)Bn * HW * HW);
  float* corr1 = A((size_t)Bn * HW * 22 * 40);
  float* corr2 = A((size_t)Bn * HW * 11 * 20);
  float* corr3 = A((size_t)Bn * HW * 5 * 10);
  float* coords= A((size_t)Bn * 2 * HW);
  float* cf    = A((size_t)Bn * 196 * HW);
  float* mc    = A((size_t)Bn * 96 * HW);
  float* flowb = A((size_t)Bn * 2 * HW);
  float* mf1   = A((size_t)Bn * 64 * HW);
  float* mf2   = A((size_t)Bn * 32 * HW);
  float* mo    = A((size_t)Bn * 80 * HW);
  float* xin   = A((size_t)Bn * 146 * HW);
  float* zb    = A((size_t)Bn * 96 * HW);
  float* rb    = A((size_t)Bn * 96 * HW);
  float* rh    = A((size_t)Bn * 96 * HW);
  float* qb    = A((size_t)Bn * 96 * HW);
  float* hh1   = A((size_t)Bn * 128 * HW);
  float* dflow = A((size_t)Bn * 2 * HW);
  // f16 packed operand buffers (sizes in halves; carve as halves/2 floats)
  _Float16* apk  = (_Float16*)A((size_t)Bn * NT * 4 * 512 / 2);   // f1^T pack
  _Float16* bpk  = (_Float16*)A((size_t)Bn * NT * 4 * 512 / 2);   // f2 pack
  _Float16* wpk  = (_Float16*)A((size_t)(96 / 16) * 7 * 512 / 2); // mc1 weights
  _Float16* cfpk = (_Float16*)A((size_t)Bn * NT * 7 * 512 / 2);   // cf pack

  const int TB = 256;
  auto blocks = [&](long long total) { return (int)((total + TB - 1) / TB); };

  auto conv = [&](const float* inA, int Ca, const float* inB, int Cb,
                  const float* wgt, const float* bias, float* out,
                  int Hh, int Ww, int Co, int K, int s, int p,
                  int OH, int OW, int act, int pre) {
    long long total = (long long)Bn * Co * OH * OW;
    conv2d_kernel<<<blocks(total), TB, 0, stream>>>(inA, Ca, inB, Cb, wgt, bias, out,
                                                    Bn, Hh, Ww, Co, K, s, p, OH, OW, act, pre);
  };

  // ---------------- feature encoder on img1 and img2 ----------------------
  conv(img1, 3, nullptr, 0, f1_w, f1_b, t1, H,  W,  32,  7, 2, 3, H2, W2, 1, 1);
  conv(t1,  32, nullptr, 0, f2_w, f2_b, t2, H2, W2, 64,  3, 2, 1, H4, W4, 1, 0);
  conv(t2,  64, nullptr, 0, f3_w, f3_b, f1f, H4, W4, 128, 3, 2, 1, h,  w,  0, 0);

  conv(img2, 3, nullptr, 0, f1_w, f1_b, t1, H,  W,  32,  7, 2, 3, H2, W2, 1, 1);
  conv(t1,  32, nullptr, 0, f2_w, f2_b, t2, H2, W2, 64,  3, 2, 1, H4, W4, 1, 0);
  conv(t2,  64, nullptr, 0, f3_w, f3_b, f2f, H4, W4, 128, 3, 2, 1, h,  w,  0, 0);

  // ---------------- context encoder on img1 -------------------------------
  conv(img1, 3, nullptr, 0, c1_w, c1_b, t1, H,  W,  32,  7, 2, 3, H2, W2, 1, 1);
  conv(t1,  32, nullptr, 0, c2_w, c2_b, t2, H2, W2, 64,  3, 2, 1, H4, W4, 1, 0);
  conv(t2,  64, nullptr, 0, c3_w, c3_b, cn, H4, W4, 160, 3, 2, 1, h,  w,  0, 0);
  split_cn_kernel<<<blocks((long long)Bn * 160 * HW), TB, 0, stream>>>(cn, hdn, ctx, Bn, HW);

  // ---------------- pack + blocked WMMA correlation + pyramid -------------
  {
    long long pt = (long long)Bn * NT * 4 * 512;
    pack_rhs_kernel<<<blocks(pt), TB, 0, stream>>>(f1f, apk, Bn, C, 4, NT, HW);
    pack_rhs_kernel<<<blocks(pt), TB, 0, stream>>>(f2f, bpk, Bn, C, 4, NT, HW);
    dim3 grid(HW / 48, HW / 48, Bn);     // 75 x 75 x 2 workgroups
    corr_gemm_kernel<<<grid, 96, 0, stream>>>(apk, bpk, corr0, HW,
                                              1.0f / sqrtf((float)C));
  }
  {
    int N = Bn * HW;
    avgpool2_kernel<<<blocks((long long)N * 22 * 40), TB, 0, stream>>>(corr0, corr1, N, 45, 80, 22, 40);
    avgpool2_kernel<<<blocks((long long)N * 11 * 20), TB, 0, stream>>>(corr1, corr2, N, 22, 40, 11, 20);
    avgpool2_kernel<<<blocks((long long)N * 5 * 10), TB, 0, stream>>>(corr2, corr3, N, 11, 20, 5, 10);
  }

  // mc1 weights -> WMMA layout once (K 196 zero-padded to 224 = 7*32)
  pack_w_kernel<<<blocks((long long)6 * 7 * 512), TB, 0, stream>>>(mc1_w, wpk, 96, 196, 7);

  set_coords_kernel<<<blocks((long long)Bn * 2 * HW), TB, 0, stream>>>(coords, Bn, h, w);

  // ---------------- 8 GRU iterations --------------------------------------
  for (int it = 0; it < ITERS; ++it) {
    corr_lookup_kernel<<<blocks((long long)Bn * 196 * HW), TB, 0, stream>>>(
        corr0, corr1, corr2, corr3, coords, cf, Bn, h, w);

    // motion encoder: 1x1 conv as WMMA GEMM (pack cf, then GEMM+bias+ReLU)
    pack_rhs_kernel<<<blocks((long long)Bn * NT * 7 * 512), TB, 0, stream>>>(
        cf, cfpk, Bn, 196, 7, NT, HW);
    {
      dim3 grid(NT, 96 / 16, Bn);        // 225 x 6 x 2 waves
      conv1x1_wmma_kernel<<<grid, 32, 0, stream>>>(wpk, cfpk, mc1_b, mc, 96, 7, HW);
    }

    build_flow_kernel<<<blocks((long long)Bn * 2 * HW), TB, 0, stream>>>(coords, flowb, Bn, h, w);
    conv(flowb, 2, nullptr, 0, mf1_w, mf1_b, mf1, h, w, 64, 7, 1, 3, h, w, 1, 0);
    conv(mf1, 64, nullptr, 0, mf2_w, mf2_b, mf2, h, w, 32, 3, 1, 1, h, w, 1, 0);
    conv(mc, 96, mf2, 32, mo_w, mo_b, mo, h, w, 80, 3, 1, 1, h, w, 1, 0);

    // xin = [ctx, mo, flow]
    build_xin_kernel<<<blocks((long long)Bn * 146 * HW), TB, 0, stream>>>(ctx, mo, coords, xin, Bn, h, w);

    // ConvGRU: hx = [hdn, xin]
    conv(hdn, 96, xin, 146, gz_w, gz_b, zb, h, w, 96, 3, 1, 1, h, w, 2, 0);
    conv(hdn, 96, xin, 146, gr_w, gr_b, rb, h, w, 96, 3, 1, 1, h, w, 2, 0);
    mul_kernel<<<blocks((long long)Bn * 96 * HW), TB, 0, stream>>>(rb, hdn, rh, Bn * 96 * HW);
    conv(rh, 96, xin, 146, gq_w, gq_b, qb, h, w, 96, 3, 1, 1, h, w, 3, 0);
    gru_update_kernel<<<blocks((long long)Bn * 96 * HW), TB, 0, stream>>>(hdn, zb, qb, Bn * 96 * HW);

    // flow head
    conv(hdn, 96, nullptr, 0, h1_w, h1_b, hh1, h, w, 128, 3, 1, 1, h, w, 1, 0);
    conv(hh1, 128, nullptr, 0, h2_w, h2_b, dflow, h, w, 2, 3, 1, 1, h, w, 0, 0);
    add_inplace_kernel<<<blocks((long long)Bn * 2 * HW), TB, 0, stream>>>(coords, dflow, Bn * 2 * HW);
  }

  // ---------------- upsample flow x8 and scale by 8 ------------------------
  upsample_flow_kernel<<<blocks((long long)Bn * 2 * H * W), TB, 0, stream>>>(
      coords, (float*)d_out, Bn, h, w, H, W);
}
